// MultiHeadAttention_4277787427432
// MI455X (gfx1250) — matrix operand order repaired and checked by reading, not compiled, not run on hardware
//
#include <hip/hip_runtime.h>
#include <math.h>

// ---------------------------------------------------------------------------
// MultiHeadAttention for MI455X (gfx1250, wave32, WMMA).
// fp32 reference emulated with bf16x3 split products on v_wmma_f32_16x16x32_bf16:
//   x = hi + lo  (hi = bf16(x), lo = bf16(x - hi))
//   A*B ~= Ah*Bh + Ah*Bl + Al*Bh        (drop lo*lo, ~fp32 accuracy)
// Pipeline: convert -> qkv GEMM (V stored transposed) -> flash attention -> out GEMM.
// ---------------------------------------------------------------------------

#define B_DIM 2
#define T_DIM 2048
#define E_DIM 1024
#define H_DIM 16
#define D_DIM 64

typedef __attribute__((ext_vector_type(16))) __bf16 v16bf;
typedef __attribute__((ext_vector_type(8)))  float  v8f;

union V16U { v16bf bf; unsigned int u[8]; };

__device__ __forceinline__ unsigned short f2bf(float f) {
  unsigned int u = __float_as_uint(f);
  u += 0x7FFFu + ((u >> 16) & 1u);          // round-to-nearest-even
  return (unsigned short)(u >> 16);
}
__device__ __forceinline__ float bf2f(unsigned short h) {
  return __uint_as_float(((unsigned int)h) << 16);
}
__device__ __forceinline__ v8f v8f_zero() {
  v8f z = {0.f, 0.f, 0.f, 0.f, 0.f, 0.f, 0.f, 0.f};
  return z;
}

// A fragment (16xK tile, K-step 32): lane = M row; half 0: K=kk0+{0..7,16..23},
// half 1: K=kk0+{8..15,24..31}.  Row-major source -> packed dword loads.
// 'off' is a 32-bit element offset of this lane's row start.
__device__ __forceinline__ v16bf load_frag_A(const unsigned short* __restrict__ base,
                                             unsigned int off, int kk0, int half) {
  V16U f;
#pragma unroll
  for (int e2 = 0; e2 < 8; ++e2) {
    int e = e2 * 2;
    int k = kk0 + e + ((e >= 8) ? 8 : 0) + half * 8;
    f.u[e2] = *(const unsigned int*)(base + off + k);
  }
  return f.bf;
}

// B fragment (Kx16 tile, K-step 32), source row-major along K:
// lane = N col; half 0: K=kk0+{0..7,16..23}, half 1: K=kk0+{8..15,24..31}.
__device__ __forceinline__ v16bf load_frag_B(const unsigned short* __restrict__ base,
                                             unsigned int off, int kk0, int half) {
  V16U f;
#pragma unroll
  for (int e2 = 0; e2 < 8; ++e2) {
    int k = kk0 + e2 * 2 + ((e2 >= 4) ? 8 : 0) + half * 8;
    f.u[e2] = *(const unsigned int*)(base + off + k);
  }
  return f.bf;
}

// bf16x3 compensated product accumulate.
__device__ __forceinline__ v8f wmma3(v16bf ah, v16bf al, v16bf bh, v16bf bl, v8f c) {
  c = __builtin_amdgcn_wmma_f32_16x16x32_bf16(false, ah, false, bh, (short)0, c, false, false);
  c = __builtin_amdgcn_wmma_f32_16x16x32_bf16(false, ah, false, bl, (short)0, c, false, false);
  c = __builtin_amdgcn_wmma_f32_16x16x32_bf16(false, al, false, bh, (short)0, c, false, false);
  return c;
}

// ---------------------------------------------------------------------------
// Kernel 0: fp32 -> (hi, lo) bf16 split
// ---------------------------------------------------------------------------
__global__ void convert_split_kernel(const float* __restrict__ src,
                                     unsigned short* __restrict__ hi,
                                     unsigned short* __restrict__ lo, int n) {
  int i = blockIdx.x * blockDim.x + threadIdx.x;
  if (i < n) {
    float f = src[i];
    unsigned short h = f2bf(f);
    hi[i] = h;
    lo[i] = f2bf(f - bf2f(h));
  }
}

// ---------------------------------------------------------------------------
// Shared GEMM core: one wave computes a 16(M) x 64(N) tile of A[M,K] @ W[N,K]^T
// ---------------------------------------------------------------------------
__device__ __forceinline__ void gemm_16x64_bf16x3(
    const unsigned short* __restrict__ Ah, const unsigned short* __restrict__ Al,
    const unsigned short* __restrict__ Wh, const unsigned short* __restrict__ Wl,
    int K, int m0, int n0, v8f acc[4]) {
  int lane = threadIdx.x & 31;
  int r = lane & 15, half = lane >> 4;
  unsigned int aoff = (unsigned int)(m0 + r) * (unsigned int)K;
  unsigned int woff[4];
#pragma unroll
  for (int ns = 0; ns < 4; ++ns)
    woff[ns] = (unsigned int)(n0 + ns * 16 + r) * (unsigned int)K;

#pragma unroll 2
  for (int kk0 = 0; kk0 < K; kk0 += 32) {
    v16bf a_h = load_frag_A(Ah, aoff, kk0, half);
    v16bf a_l = load_frag_A(Al, aoff, kk0, half);
#pragma unroll
    for (int ns = 0; ns < 4; ++ns) {
      v16bf b_h = load_frag_B(Wh, woff[ns], kk0, half);
      v16bf b_l = load_frag_B(Wl, woff[ns], kk0, half);
      acc[ns] = wmma3(a_h, a_l, b_h, b_l, acc[ns]);
    }
  }
}

// ---------------------------------------------------------------------------
// Kernel 1: qkv = x @ Wqkv^T, scattered into Q/K [b,h,t,d] and V [b,h,d,t]
// (V transposed so the PV GEMM's B fragments are contiguous in K=j).
// Column c of qkv decodes as c = d*48 + which*16 + head (reference reshape).
// Each 16-wide N-subtile has uniform (d, which); head == lane's N index.
// ---------------------------------------------------------------------------
__global__ void qkv_proj_kernel(const unsigned short* __restrict__ xh,
                                const unsigned short* __restrict__ xl,
                                const unsigned short* __restrict__ wh,
                                const unsigned short* __restrict__ wl,
                                unsigned short* __restrict__ qh, unsigned short* __restrict__ ql,
                                unsigned short* __restrict__ kh, unsigned short* __restrict__ kl,
                                unsigned short* __restrict__ vh, unsigned short* __restrict__ vl) {
  int m0 = blockIdx.x * 16;   // row tile over b*t = 4096
  int n0 = blockIdx.y * 64;   // col tile over 3*E = 3072
  v8f acc[4];
#pragma unroll
  for (int i = 0; i < 4; ++i) acc[i] = v8f_zero();

  gemm_16x64_bf16x3(xh, xl, wh, wl, E_DIM, m0, n0, acc);

  int lane = threadIdx.x & 31;
  int nn = lane & 15, half = lane >> 4;
#pragma unroll
  for (int ns = 0; ns < 4; ++ns) {
    int c0 = n0 + ns * 16;
    int dd = c0 / 48;
    int ki = (c0 % 48) >> 4;              // 0=Q 1=K 2=V
    unsigned short* dh = (ki == 0) ? qh : (ki == 1) ? kh : vh;
    unsigned short* dl = (ki == 0) ? ql : (ki == 1) ? kl : vl;
#pragma unroll
    for (int i = 0; i < 8; ++i) {
      int row = m0 + i + 8 * half;        // global b*t row
      int b   = row >> 11;                // /2048
      int tok = row & (T_DIM - 1);
      size_t idx;
      if (ki == 2) {                      // V: [b,h,d,t]
        idx = (((size_t)(b * H_DIM + nn)) * D_DIM + dd) * T_DIM + tok;
      } else {                            // Q,K: [b,h,t,d]
        idx = (((size_t)(b * H_DIM + nn)) * T_DIM + tok) * D_DIM + dd;
      }
      float f = acc[ns][i];
      unsigned short hv = f2bf(f);
      dh[idx] = hv;
      dl[idx] = f2bf(f - bf2f(hv));
    }
  }
}

// ---------------------------------------------------------------------------
// Kernel 2: flash attention.  One wave per (b*h, 16-row q tile); streams 32
// keys/values per step; online softmax; P transposed C->A layout via LDS.
// V is pre-transposed [b,h,d,t] so its B fragments are packed dword loads.
// ---------------------------------------------------------------------------
__global__ void attention_kernel(const unsigned short* __restrict__ qh,
                                 const unsigned short* __restrict__ ql,
                                 const unsigned short* __restrict__ kh,
                                 const unsigned short* __restrict__ kl,
                                 const unsigned short* __restrict__ vh,
                                 const unsigned short* __restrict__ vl,
                                 unsigned short* __restrict__ oh,
                                 unsigned short* __restrict__ ol) {
  __shared__ float pt[16][33];            // 16x32 P tile + pad

  int qt   = blockIdx.x;                  // 0..127 q-row tile
  int bh   = blockIdx.y;                  // 0..31  (b*16 + h)
  int lane = threadIdx.x & 31;
  int r = lane & 15, half = lane >> 4;

  const unsigned int base = (unsigned int)bh * T_DIM * D_DIM;

  // Q fragments (16 rows x d=64): two K-steps, hi & lo.
  unsigned int qoff = base + (unsigned int)(qt * 16 + r) * D_DIM;
  v16bf Qhf[2], Qlf[2];
  Qhf[0] = load_frag_A(qh, qoff, 0, half);  Qhf[1] = load_frag_A(qh, qoff, 32, half);
  Qlf[0] = load_frag_A(ql, qoff, 0, half);  Qlf[1] = load_frag_A(ql, qoff, 32, half);

  // V (transposed) row offsets: lane's row = d-column dt*16 + r, stride T.
  unsigned int voff[4];
#pragma unroll
  for (int dt = 0; dt < 4; ++dt)
    voff[dt] = base + (unsigned int)(dt * 16 + r) * T_DIM;

  v8f acc[4];
  float m_run[8], l_run[8];
#pragma unroll
  for (int i = 0; i < 4; ++i) acc[i] = v8f_zero();
#pragma unroll
  for (int i = 0; i < 8; ++i) { m_run[i] = -INFINITY; l_run[i] = 0.f; }

  for (int j0 = 0; j0 < T_DIM; j0 += 32) {
    // ---- S = Q @ K^T (scaled), two 16x16 column tiles -------------------
    v8f s[2] = {v8f_zero(), v8f_zero()};
#pragma unroll
    for (int nt = 0; nt < 2; ++nt) {
      unsigned int koff = base + (unsigned int)(j0 + nt * 16 + r) * D_DIM;
#pragma unroll
      for (int ks = 0; ks < 2; ++ks) {
        v16bf b_h = load_frag_B(kh, koff, ks * 32, half);
        v16bf b_l = load_frag_B(kl, koff, ks * 32, half);
        s[nt] = wmma3(Qhf[ks], Qlf[ks], b_h, b_l, s[nt]);
      }
    }

    // ---- online softmax (rows live in 16-lane halves) -------------------
#pragma unroll
    for (int i = 0; i < 8; ++i) {
      float s0 = s[0][i] * 0.125f;        // 1/sqrt(64)
      float s1 = s[1][i] * 0.125f;
      float rm = fmaxf(s0, s1);
#pragma unroll
      for (int msk = 1; msk <= 8; msk <<= 1) rm = fmaxf(rm, __shfl_xor(rm, msk, 32));
      float nm    = fmaxf(m_run[i], rm);
      float alpha = __expf(m_run[i] - nm);
      float p0 = __expf(s0 - nm);
      float p1 = __expf(s1 - nm);
      float rs = p0 + p1;
#pragma unroll
      for (int msk = 1; msk <= 8; msk <<= 1) rs += __shfl_xor(rs, msk, 32);
      l_run[i] = l_run[i] * alpha + rs;
      m_run[i] = nm;
      acc[0][i] *= alpha; acc[1][i] *= alpha; acc[2][i] *= alpha; acc[3][i] *= alpha;
      pt[i + 8 * half][r]      = p0;      // C layout -> LDS
      pt[i + 8 * half][16 + r] = p1;
    }
    __syncthreads();

    // ---- P: C layout -> A layout (16x32, K=32), split hi/lo -------------
    V16U fh, fl;
#pragma unroll
    for (int e2 = 0; e2 < 8; ++e2) {
      int e = e2 * 2;
      int k = e + ((e >= 8) ? 8 : 0) + half * 8;
      float f0 = pt[r][k], f1 = pt[r][k + 1];
      unsigned short h0 = f2bf(f0), h1 = f2bf(f1);
      unsigned short l0 = f2bf(f0 - bf2f(h0)), l1 = f2bf(f1 - bf2f(h1));
      fh.u[e2] = (unsigned)h0 | ((unsigned)h1 << 16);
      fl.u[e2] = (unsigned)l0 | ((unsigned)l1 << 16);
    }
    __syncthreads();

    // ---- O += P @ V (4 d-tiles of 16); V^T rows contiguous in j ---------
#pragma unroll
    for (int dt = 0; dt < 4; ++dt) {
      v16bf b_h = load_frag_B(vh, voff[dt], j0, half);
      v16bf b_l = load_frag_B(vl, voff[dt], j0, half);
      acc[dt] = wmma3(fh.bf, fl.bf, b_h, b_l, acc[dt]);
    }
  }

  // ---- epilogue: O /= l, store hi/lo bf16 in [b, t, h*64+d] ---------------
  int b = bh >> 4, hd = bh & 15;
#pragma unroll
  for (int i = 0; i < 8; ++i) {
    float inv = 1.0f / l_run[i];
    int tok = qt * 16 + i + 8 * half;
#pragma unroll
    for (int dt = 0; dt < 4; ++dt) {
      float o = acc[dt][i] * inv;
      size_t idx = ((size_t)b * T_DIM + tok) * E_DIM + hd * 64 + dt * 16 + r;
      unsigned short hv = f2bf(o);
      oh[idx] = hv;
      ol[idx] = f2bf(o - bf2f(hv));
    }
  }
}

// ---------------------------------------------------------------------------
// Kernel 3: out = attn @ Wout^T, fp32 epilogue straight to d_out.
// ---------------------------------------------------------------------------
__global__ void out_proj_kernel(const unsigned short* __restrict__ ah,
                                const unsigned short* __restrict__ al,
                                const unsigned short* __restrict__ wh,
                                const unsigned short* __restrict__ wl,
                                float* __restrict__ out) {
  int m0 = blockIdx.x * 16;
  int n0 = blockIdx.y * 64;
  v8f acc[4];
#pragma unroll
  for (int i = 0; i < 4; ++i) acc[i] = v8f_zero();

  gemm_16x64_bf16x3(ah, al, wh, wl, E_DIM, m0, n0, acc);

  int lane = threadIdx.x & 31;
  int nn = lane & 15, half = lane >> 4;
#pragma unroll
  for (int ns = 0; ns < 4; ++ns)
#pragma unroll
    for (int i = 0; i < 8; ++i)
      out[(size_t)(m0 + i + 8 * half) * E_DIM + n0 + ns * 16 + nn] = acc[ns][i];
}

// ---------------------------------------------------------------------------
// Host launch
// ---------------------------------------------------------------------------
extern "C" void kernel_launch(void* const* d_in, const int* in_sizes, int n_in,
                              void* d_out, int out_size, void* d_ws, size_t ws_size,
                              hipStream_t stream) {
  (void)in_sizes; (void)n_in; (void)out_size; (void)ws_size;
  const float* x    = (const float*)d_in[0];
  const float* wqkv = (const float*)d_in[1];
  const float* wout = (const float*)d_in[2];
  float* out = (float*)d_out;

  unsigned short* ws = (unsigned short*)d_ws;
  size_t off = 0;
  auto alloc = [&](size_t n) { unsigned short* p = ws + off; off += n; return p; };

  const size_t NX   = (size_t)B_DIM * T_DIM * E_DIM;          // 4M
  const size_t NWQ  = (size_t)3 * E_DIM * E_DIM;              // 3M
  const size_t NWO  = (size_t)E_DIM * E_DIM;                  // 1M
  const size_t NQKV = (size_t)B_DIM * H_DIM * T_DIM * D_DIM;  // 4M

  unsigned short *xh = alloc(NX),  *xl = alloc(NX);
  unsigned short *wqh = alloc(NWQ), *wql = alloc(NWQ);
  unsigned short *woh = alloc(NWO), *wol = alloc(NWO);
  unsigned short *qh = alloc(NQKV), *ql = alloc(NQKV);
  unsigned short *kh = alloc(NQKV), *kl = alloc(NQKV);
  unsigned short *vh = alloc(NQKV), *vl = alloc(NQKV);
  unsigned short *ah = alloc(NX),  *al = alloc(NX);

  convert_split_kernel<<<dim3((unsigned)((NX  + 255) / 256)), 256, 0, stream>>>(x,    xh,  xl,  (int)NX);
  convert_split_kernel<<<dim3((unsigned)((NWQ + 255) / 256)), 256, 0, stream>>>(wqkv, wqh, wql, (int)NWQ);
  convert_split_kernel<<<dim3((unsigned)((NWO + 255) / 256)), 256, 0, stream>>>(wout, woh, wol, (int)NWO);

  // M=4096 -> 256 row tiles; N=3072 -> 48 groups of 64
  qkv_proj_kernel<<<dim3(256, 48), 32, 0, stream>>>(xh, xl, wqh, wql, qh, ql, kh, kl, vh, vl);
  // 128 q-tiles x 32 (b*h)
  attention_kernel<<<dim3(128, 32), 32, 0, stream>>>(qh, ql, kh, kl, vh, vl, ah, al);
  // M=4096 -> 256; N=1024 -> 16 groups of 64
  out_proj_kernel<<<dim3(256, 16), 32, 0, stream>>>(ah, al, woh, wol, out);
}
